// BasicMultiHeadAttention_68753836474599
// MI455X (gfx1250) — matrix operand order repaired and checked by reading, not compiled, not run on hardware
//
#include <hip/hip_runtime.h>
#include <hip/hip_bf16.h>

// ---------------- CDNA5 WMMA types ----------------
typedef __attribute__((ext_vector_type(16))) __bf16        v16bf;
typedef __attribute__((ext_vector_type(8)))  float         v8f;
typedef __attribute__((ext_vector_type(8)))  unsigned int  v8u;
typedef __attribute__((ext_vector_type(4)))  float         f32x4;

static constexpr int Bn = 4;
static constexpr int Sn = 2048;
static constexpr int Dm = 1024;
static constexpr int Hn = 16;
static constexpr int Dk = 64;

__device__ __forceinline__ v8f wmma_bf16(v16bf a, v16bf b, v8f c) {
  // D = A(16x32 bf16) x B(32x16 bf16) + C(16x16 f32)
  return __builtin_amdgcn_wmma_f32_16x16x32_bf16(
      /*neg_a=*/false, a, /*neg_b=*/false, b,
      /*c_mod=*/(short)0, c, /*reuse_a=*/false, /*reuse_b=*/false);
}

__device__ __forceinline__ unsigned short f2bf_bits(float f) {
  return __builtin_bit_cast(unsigned short, (__bf16)f);
}

// ---- A fragment (16x32, 16-bit) from row-major bf16, row stride ld elems ----
// lane lo=lane&15 -> row M=row0+lo ; elems 0..7 -> K = hi*8 + 0..7 ; 8..15 -> +16
__device__ __forceinline__ v16bf load_a_bf16(const unsigned short* base,
                                             int row0, int k0, int ld, int lane) {
  const int lo = lane & 15, hi = lane >> 4;
  const unsigned short* p = base + (size_t)(row0 + lo) * ld + k0 + hi * 8;
  uint4 x = *(const uint4*)(p);        // K offsets 0..7   (2 bf16 per dword)
  uint4 y = *(const uint4*)(p + 16);   // K offsets 16..23
  v8u raw;
  raw[0] = x.x; raw[1] = x.y; raw[2] = x.z; raw[3] = x.w;
  raw[4] = y.x; raw[5] = y.y; raw[6] = y.z; raw[7] = y.w;
  return __builtin_bit_cast(v16bf, raw);
}

// ---- A fragment (16x32) from row-major f32 with convert-on-load (global or LDS) ----
__device__ __forceinline__ v16bf load_a_f32(const float* base,
                                            int row0, int k0, int ld, int lane) {
  const int lo = lane & 15, hi = lane >> 4;
  const float* p = base + (size_t)(row0 + lo) * ld + k0 + hi * 8;
  f32x4 f0 = *(const f32x4*)(p);
  f32x4 f1 = *(const f32x4*)(p + 4);
  f32x4 f2 = *(const f32x4*)(p + 16);
  f32x4 f3 = *(const f32x4*)(p + 20);
  v16bf a;
#pragma unroll
  for (int i = 0; i < 4; ++i) {
    a[i]      = (__bf16)f0[i];
    a[4 + i]  = (__bf16)f1[i];
    a[8 + i]  = (__bf16)f2[i];
    a[12 + i] = (__bf16)f3[i];
  }
  return a;
}

// ---- B fragment (32x16) where B[k][n] = S[n][k], S row-major bf16 stride ld ----
// lane: N = n0 + (lane&15) ; elems 0..7 -> K = k0 + hi*8 + 0..7 ; 8..15 -> +16
__device__ __forceinline__ v16bf load_bT_bf16(const unsigned short* base,
                                              int n0, int k0, int ld, int lane) {
  const int lo = lane & 15, hi = lane >> 4;
  const unsigned short* p = base + (size_t)(n0 + lo) * ld + k0 + hi * 8;
  uint4 x = *(const uint4*)(p);
  uint4 y = *(const uint4*)(p + 16);
  v8u raw;
  raw[0] = x.x; raw[1] = x.y; raw[2] = x.z; raw[3] = x.w;
  raw[4] = y.x; raw[5] = y.y; raw[6] = y.z; raw[7] = y.w;
  return __builtin_bit_cast(v16bf, raw);
}

// ---- C/D tile stores (VGPR r -> M = hi*8+r, N = lane&15) ----
__device__ __forceinline__ void store_tile_bf16(unsigned short* Y, int m0, int n0,
                                                int ld, v8f acc, int lane) {
  const int lo = lane & 15, hi = lane >> 4;
#pragma unroll
  for (int r = 0; r < 8; ++r)
    Y[(size_t)(m0 + hi * 8 + r) * ld + n0 + lo] = f2bf_bits(acc[r]);
}

__device__ __forceinline__ void store_tile_f32(float* Y, int m0, int n0,
                                               int ld, v8f acc, int lane) {
  const int lo = lane & 15, hi = lane >> 4;
#pragma unroll
  for (int r = 0; r < 8; ++r)
    Y[(size_t)(m0 + hi * 8 + r) * ld + n0 + lo] = acc[r];
}

// transposed store into Vt[b,h,dk,s]: tile row n -> Vt row, 8 contiguous s per lane
__device__ __forceinline__ void store_tile_vt(unsigned short* Y, int m0, int n0,
                                              v8f acc, int lane) {
  const int lo = lane & 15, hi = lane >> 4;
  const int b = m0 / Sn, s0 = m0 % Sn;       // 16-row tile never crosses batch
  const int h = n0 / Dk, dk = n0 % Dk;
  unsigned short u[8];
#pragma unroll
  for (int r = 0; r < 8; ++r) u[r] = f2bf_bits(acc[r]);
  uint4 o;
  o.x = (unsigned)u[0] | ((unsigned)u[1] << 16);
  o.y = (unsigned)u[2] | ((unsigned)u[3] << 16);
  o.z = (unsigned)u[4] | ((unsigned)u[5] << 16);
  o.w = (unsigned)u[6] | ((unsigned)u[7] << 16);
  size_t row = (size_t)((b * Hn + h) * Dk + dk + lo);
  *(uint4*)(Y + row * Sn + s0 + hi * 8) = o;
}

// =============== Kernel 0: one-shot f32 -> bf16 weight conversion ===============
__global__ __launch_bounds__(256) void cvt_kernel(const float* __restrict__ src,
                                                  unsigned short* __restrict__ dst,
                                                  int n4) {
  const int i = blockIdx.x * blockDim.x + threadIdx.x;
  if (i >= n4) return;
  f32x4 f = ((const f32x4*)src)[i];
  uint2 o;
  o.x = (unsigned)f2bf_bits(f[0]) | ((unsigned)f2bf_bits(f[1]) << 16);
  o.y = (unsigned)f2bf_bits(f[2]) | ((unsigned)f2bf_bits(f[3]) << 16);
  ((uint2*)dst)[i] = o;
}

// ================= Kernel 1: per-head projections  Y = X @ W^T =================
// 2x2 register blocking: one wave32 computes a 32x32 output block (4 accumulators).
template <bool TRANS_OUT>
__global__ __launch_bounds__(128) void proj_kernel(const float* __restrict__ X,
                                                   const unsigned short* __restrict__ Wb,
                                                   unsigned short* __restrict__ Y) {
  const int lane = threadIdx.x & 31;
  const int wid  = blockIdx.x * (blockDim.x >> 5) + (threadIdx.x >> 5);
  const int nt   = Dm / 32;                   // 32 N-blocks
  const int m0   = (wid / nt) * 32;
  const int n0   = (wid % nt) * 32;

  v8f c00 = {}, c01 = {}, c10 = {}, c11 = {};
#pragma unroll 2
  for (int k0 = 0; k0 < Dm; k0 += 32) {
    v16bf a0 = load_a_f32(X, m0,      k0, Dm, lane);
    v16bf a1 = load_a_f32(X, m0 + 16, k0, Dm, lane);
    v16bf b0 = load_bT_bf16(Wb, n0,      k0, Dm, lane);
    v16bf b1 = load_bT_bf16(Wb, n0 + 16, k0, Dm, lane);
    c00 = wmma_bf16(a0, b0, c00);
    c01 = wmma_bf16(a0, b1, c01);
    c10 = wmma_bf16(a1, b0, c10);
    c11 = wmma_bf16(a1, b1, c11);
  }

  if (!TRANS_OUT) {
    store_tile_bf16(Y, m0,      n0,      Dm, c00, lane);
    store_tile_bf16(Y, m0,      n0 + 16, Dm, c01, lane);
    store_tile_bf16(Y, m0 + 16, n0,      Dm, c10, lane);
    store_tile_bf16(Y, m0 + 16, n0 + 16, Dm, c11, lane);
  } else {
    store_tile_vt(Y, m0,      n0,      c00, lane);
    store_tile_vt(Y, m0,      n0 + 16, c01, lane);
    store_tile_vt(Y, m0 + 16, n0,      c10, lane);
    store_tile_vt(Y, m0 + 16, n0 + 16, c11, lane);
  }
}

// ==== Kernel 2 (fused): scores -> softmax -> attn (NT store) -> P@V -> concat ====
// One workgroup (8 wave32) per 16-query block of one (b,h).
// Dynamic LDS: 16x2048 f32 exp strip (128KB) + 16 f32 rowscale + 4x16x16 f32 partials.
// attn is written once with non-temporal stores and never re-read (saves 1.07 GB HBM).
__global__ __launch_bounds__(256) void scores_softmax_av_kernel(
    const unsigned short* __restrict__ Qb, const unsigned short* __restrict__ Kb,
    const unsigned short* __restrict__ Vt, float* __restrict__ attn,
    unsigned short* __restrict__ concat) {
  extern __shared__ float smem[];
  float* sc   = smem;                          // [16][2048] scores / exp
  float* rinv = smem + 16 * Sn;                // [16] 1/rowsum
  float* pred = rinv + 16;                     // [4][16][16] K-half partials

  const int lane = threadIdx.x & 31;
  const int wave = threadIdx.x >> 5;          // 0..7
  const int qt = blockIdx.x & (Sn / 16 - 1);  // 0..127
  const int bh = blockIdx.x >> 7;             // 0..63
  const int h = bh & (Hn - 1), b = bh >> 4;
  const int q0 = qt * 16;
  const int lo = lane & 15, hi = lane >> 4;

  const unsigned short* Qp = Qb + (size_t)b * Sn * Dm + (size_t)h * Dk; // ld=Dm
  const unsigned short* Kp = Kb + (size_t)b * Sn * Dm + (size_t)h * Dk;
  const unsigned short* Vp = Vt + (size_t)(b * Hn + h) * Dk * Sn;       // rows=dk, ld=Sn

  // ---------------- Phase 1: scores strip (Q frags resident) ----------------
  v16bf a0 = load_a_bf16(Qp, q0, 0,  Dm, lane);
  v16bf a1 = load_a_bf16(Qp, q0, 32, Dm, lane);

#pragma unroll 2
  for (int t = 0; t < 16; ++t) {              // wave covers 256 key columns
    const int n0 = wave * 256 + t * 16;
    v16bf b0 = load_bT_bf16(Kp, n0, 0,  Dm, lane);
    v16bf b1 = load_bT_bf16(Kp, n0, 32, Dm, lane);
    v8f acc = {};
    acc = wmma_bf16(a0, b0, acc);
    acc = wmma_bf16(a1, b1, acc);
#pragma unroll
    for (int r = 0; r < 8; ++r)
      sc[(hi * 8 + r) * Sn + n0 + lo] = acc[r] * 0.125f;   // 1/sqrt(64)
  }
  __syncthreads();

  // ------------- Phase 2: softmax (2 rows/wave) + NT attn write -------------
#pragma unroll
  for (int rr = 0; rr < 2; ++rr) {
    const int m = wave * 2 + rr;
    float* row = sc + (size_t)m * Sn;
    float mx = -3.4e38f;
    for (int i = lane; i < Sn; i += 32) mx = fmaxf(mx, row[i]);
#pragma unroll
    for (int off = 16; off > 0; off >>= 1) mx = fmaxf(mx, __shfl_xor(mx, off, 32));
    float sum = 0.f;
    for (int i = lane; i < Sn; i += 32) {
      float e = __expf(row[i] - mx);
      row[i] = e;                              // keep un-normalized exp in LDS
      sum += e;
    }
#pragma unroll
    for (int off = 16; off > 0; off >>= 1) sum += __shfl_xor(sum, off, 32);
    const float inv = 1.0f / sum;
    if (lane == 0) rinv[m] = inv;
    f32x4* dst = (f32x4*)(attn + ((size_t)(b * Hn + h) * Sn + q0 + m) * Sn);
    const f32x4* src = (const f32x4*)row;
    for (int i = lane; i < Sn / 4; i += 32) {
      f32x4 v = src[i] * inv;
      __builtin_nontemporal_store(v, dst + i);  // attn never re-read: bypass caches
    }
  }
  __syncthreads();

  // ------- Phase 3: O = P~ @ V from LDS; wave = (dk-tile, K-half) -------
  const int nt = wave & 3;                     // dk tile: n0 = nt*16
  const int kh = wave >> 2;                    // K half: [kh*1024, kh*1024+1024)
  const int n0 = nt * 16;
  v8f acc = {};
#pragma unroll 2
  for (int ks = 0; ks < Sn / 2; ks += 32) {
    const int k0 = kh * (Sn / 2) + ks;
    v16bf a  = load_a_f32(sc, 0, k0, Sn, lane);        // LDS exp -> bf16 frags
    v16bf bb = load_bT_bf16(Vp, n0, k0, Sn, lane);
    acc = wmma_bf16(a, bb, acc);
  }
  __syncthreads();
  if (kh == 1) {
#pragma unroll
    for (int r = 0; r < 8; ++r)
      pred[(nt * 16 + hi * 8 + r) * 16 + lo] = acc[r];
  }
  __syncthreads();
  if (kh == 0) {
    unsigned short* Cp = concat + (size_t)b * Sn * Dm + h * Dk;  // ld = Dm
#pragma unroll
    for (int r = 0; r < 8; ++r) {
      float v = (acc[r] + pred[(nt * 16 + hi * 8 + r) * 16 + lo]) * rinv[hi * 8 + r];
      Cp[(size_t)(q0 + hi * 8 + r) * Dm + n0 + lo] = f2bf_bits(v);
    }
  }
}

// ================= Kernel 3: output = concat @ Wo^T  (f32 out) =================
__global__ __launch_bounds__(128) void outproj_kernel(
    const unsigned short* __restrict__ Cc, const unsigned short* __restrict__ Wob,
    float* __restrict__ out) {
  const int lane = threadIdx.x & 31;
  const int wid  = blockIdx.x * (blockDim.x >> 5) + (threadIdx.x >> 5);
  const int nt   = Dm / 32;
  const int m0   = (wid / nt) * 32;
  const int n0   = (wid % nt) * 32;

  v8f c00 = {}, c01 = {}, c10 = {}, c11 = {};
#pragma unroll 2
  for (int k0 = 0; k0 < Dm; k0 += 32) {
    v16bf a0 = load_a_bf16(Cc, m0,      k0, Dm, lane);
    v16bf a1 = load_a_bf16(Cc, m0 + 16, k0, Dm, lane);
    v16bf b0 = load_bT_bf16(Wob, n0,      k0, Dm, lane);
    v16bf b1 = load_bT_bf16(Wob, n0 + 16, k0, Dm, lane);
    c00 = wmma_bf16(a0, b0, c00);
    c01 = wmma_bf16(a0, b1, c01);
    c10 = wmma_bf16(a1, b0, c10);
    c11 = wmma_bf16(a1, b1, c11);
  }

  store_tile_f32(out, m0,      n0,      Dm, c00, lane);
  store_tile_f32(out, m0,      n0 + 16, Dm, c01, lane);
  store_tile_f32(out, m0 + 16, n0,      Dm, c10, lane);
  store_tile_f32(out, m0 + 16, n0 + 16, Dm, c11, lane);
}

// ================================ launch ================================
extern "C" void kernel_launch(void* const* d_in, const int* in_sizes, int n_in,
                              void* d_out, int out_size, void* d_ws, size_t ws_size,
                              hipStream_t stream) {
  const float* query = (const float*)d_in[0];   // [4,2048,1024]
  const float* key   = (const float*)d_in[1];
  const float* value = (const float*)d_in[2];
  const float* Wq    = (const float*)d_in[3];   // [16,64,1024] == [1024,1024]
  const float* Wk    = (const float*)d_in[4];
  const float* Wv    = (const float*)d_in[5];
  const float* Wo    = (const float*)d_in[6];   // [1024,1024]

  float* out  = (float*)d_out;                         // [4,2048,1024]
  float* attn = out + (size_t)Bn * Sn * Dm;            // [4,16,2048,2048]

  const size_t welems = (size_t)Dm * Dm;               // 1M bf16 per weight
  const size_t elems  = (size_t)Bn * Sn * Dm;          // 8M bf16 each
  unsigned short* Wqb = (unsigned short*)d_ws;
  unsigned short* Wkb = Wqb + welems;
  unsigned short* Wvb = Wkb + welems;
  unsigned short* Wob = Wvb + welems;
  unsigned short* Qb  = Wob + welems;
  unsigned short* Kb  = Qb + elems;
  unsigned short* Vt  = Kb + elems;                    // [Bn,Hn,Dk,Sn]
  unsigned short* Cc  = Vt + elems;                    // concat bf16

  // Weight conversion: 1M elems each -> 262144 float4 threads
  const int n4 = (int)(welems / 4);
  cvt_kernel<<<(n4 + 255) / 256, 256, 0, stream>>>(Wq, Wqb, n4);
  cvt_kernel<<<(n4 + 255) / 256, 256, 0, stream>>>(Wk, Wkb, n4);
  cvt_kernel<<<(n4 + 255) / 256, 256, 0, stream>>>(Wv, Wvb, n4);
  cvt_kernel<<<(n4 + 255) / 256, 256, 0, stream>>>(Wo, Wob, n4);

  // 256 M-blocks x 32 N-blocks = 8192 waves ; 4 waves/block -> 2048 blocks
  proj_kernel<false><<<2048, 128, 0, stream>>>(query, Wqb, Qb);
  proj_kernel<false><<<2048, 128, 0, stream>>>(key,   Wkb, Kb);
  proj_kernel<true ><<<2048, 128, 0, stream>>>(value, Wvb, Vt);

  // 64 (b,h) x 128 q-tiles = 8192 groups; LDS = 128KB strip + 16 + 1024 floats
  const size_t smem_bytes = (size_t)(16 * Sn + 16 + 4 * 16 * 16) * sizeof(float);
  scores_softmax_av_kernel<<<8192, 256, smem_bytes, stream>>>(Qb, Kb, Vt, attn, Cc);

  outproj_kernel<<<2048, 128, 0, stream>>>(Cc, Wob, out);
}
